// MultiHeadAttention_54631984005437
// MI455X (gfx1250) — matrix operand order repaired and checked by reading, not compiled, not run on hardware
//
#include <hip/hip_runtime.h>

// ---------------------------------------------------------------------------
// MultiHeadAttention for MI455X (gfx1250): bf16 WMMA everywhere, fp32 accum.
// Double-buffered LDS pipelines, async global->LDS staging (ASYNCcnt),
// DPP16 softmax reductions, prefetch hints.
// ---------------------------------------------------------------------------

typedef __bf16 bf16;
typedef __attribute__((ext_vector_type(16))) __bf16 v16bf;
typedef __attribute__((ext_vector_type(8)))  __bf16 bf16x8;
typedef __attribute__((ext_vector_type(8)))  float  v8f;
typedef __attribute__((ext_vector_type(4)))  float  f32x4;

#define D_MODEL  1024
#define N_HEADS  16
#define HEAD_DIM 64
#define BATCH    4
#define SEQ      2048
#define M_TOKENS (BATCH * SEQ)   // 8192

static __device__ __forceinline__ v16bf pack16(bf16x8 lo, bf16x8 hi) {
  union { v16bf v; bf16x8 h[2]; } u;
  u.h[0] = lo; u.h[1] = hi;
  return u.v;
}

static __device__ __forceinline__ v8f wmma_bf16(v16bf a, v16bf b, v8f c) {
  // D = A(16x32) * B(32x16) + C(16x16 f32)
  return __builtin_amdgcn_wmma_f32_16x16x32_bf16(false, a, false, b,
                                                 (short)0, c, false, false);
}

// --- async global -> LDS 16B copy (per lane), tracked by ASYNCcnt ----------
static __device__ __forceinline__ void async_copy_b128(unsigned lds_off,
                                                       const void* gaddr) {
  asm volatile("global_load_async_to_lds_b128 %0, %1, off"
               :: "v"(lds_off), "v"(gaddr)
               : "memory");
}
static __device__ __forceinline__ void wait_asynccnt0() {
  asm volatile("s_wait_asynccnt 0" ::: "memory");
}
static __device__ __forceinline__ unsigned lds_addr(const void* p) {
  return (unsigned)(unsigned long long)p;   // low 32 bits of flat = LDS offset
}

// --- DPP16 butterfly reductions within each 16-lane group (VALU only) ------
#define DPP_F32(x, ctrl)                                                      \
  __builtin_bit_cast(float, __builtin_amdgcn_update_dpp(                      \
      0, __builtin_bit_cast(int, (x)), (ctrl), 0xF, 0xF, true))

static __device__ __forceinline__ float red_max16(float x) {
  x = fmaxf(x, DPP_F32(x, 0xB1));   // quad_perm [1,0,3,2]
  x = fmaxf(x, DPP_F32(x, 0x4E));   // quad_perm [2,3,0,1]
  x = fmaxf(x, DPP_F32(x, 0x141));  // row_half_mirror
  x = fmaxf(x, DPP_F32(x, 0x140));  // row_mirror
  return x;
}
static __device__ __forceinline__ float red_add16(float x) {
  x += DPP_F32(x, 0xB1);
  x += DPP_F32(x, 0x4E);
  x += DPP_F32(x, 0x141);
  x += DPP_F32(x, 0x140);
  return x;
}

// ---------------------------------------------------------------------------
// GEMM:  C[M,N] = A[M,K] @ W[N,K]^T + bias[N]
// Block tile 128x64, K-tile 64 (2 WMMA steps), double-buffered LDS.
// fp32 A: load->convert->LDS pipeline.  bf16 A: async global->LDS staging.
// ---------------------------------------------------------------------------
template <typename AT, typename OT>
__global__ __launch_bounds__(256) void gemm_bias_kernel(
    const AT* __restrict__ A, const float* __restrict__ W,
    const float* __restrict__ bias, OT* __restrict__ C,
    int M, int N, int K)
{
  constexpr int BM = 128, BN = 64, KT = 64;
  constexpr int LDA = 72, LDB = 72;          // padded halves; rows 16B-aligned
  constexpr bool A_IS_F32 = (sizeof(AT) == 4);
  __shared__ bf16 sA[2][BM * LDA];
  __shared__ bf16 sB[2][BN * LDB];

  const int tid  = threadIdx.x;
  const int lane = tid & 31;
  const int wv   = tid >> 5;
  const int wM   = wv >> 1;                  // 0..3
  const int wN   = wv & 1;                   // 0..1
  const int m0   = blockIdx.y * BM;
  const int n0   = blockIdx.x * BN;

  const int r     = lane & 15;
  const int hi    = lane >> 4;
  const int koffA = hi * 16;                 // A frag: 16 contiguous K
  const int koffB = hi * 16;                 // B frag: 16 contiguous K

  v8f acc[2][2];
  #pragma unroll
  for (int i = 0; i < 2; ++i)
    #pragma unroll
    for (int j = 0; j < 2; ++j)
      acc[i][j] = v8f{0.f, 0.f, 0.f, 0.f, 0.f, 0.f, 0.f, 0.f};

  // staging: A -> 2 threads/row * 32 elems; B -> 4 threads/row * 16 elems
  const int arow = tid >> 1, acol = (tid & 1) * 32;
  const int brow = tid >> 2, bcol = (tid & 3) * 16;
  const AT*    gA = A + (size_t)(m0 + arow) * K + acol;
  const float* gB = W + (size_t)(n0 + brow) * K + bcol;

  f32x4 raf[8];    // A tile regs (fp32 source only)
  f32x4 rb[4];     // B tile regs (always fp32)

  auto stageA_async = [&](int k0, int buf) {   // bf16 A: no VGPR round trip
    #pragma unroll
    for (int c = 0; c < 4; ++c)
      async_copy_b128(lds_addr(&sA[buf][arow * LDA + acol + c * 8]),
                      gA + k0 + c * 8);
  };
  auto stage_load = [&](int k0) {
    if constexpr (A_IS_F32) {
      #pragma unroll
      for (int i = 0; i < 8; ++i) raf[i] = *(const f32x4*)(gA + k0 + i * 4);
    }
    #pragma unroll
    for (int i = 0; i < 4; ++i) rb[i] = *(const f32x4*)(gB + k0 + i * 4);
  };
  auto stage_store = [&](int buf) {
    if constexpr (A_IS_F32) {
      bf16* lA = &sA[buf][arow * LDA + acol];
      #pragma unroll
      for (int i = 0; i < 4; ++i) {
        bf16x8 b;
        #pragma unroll
        for (int e = 0; e < 4; ++e) {
          b[e] = (bf16)raf[2 * i][e]; b[4 + e] = (bf16)raf[2 * i + 1][e];
        }
        *(bf16x8*)(lA + i * 8) = b;
      }
    }
    bf16* lB = &sB[buf][brow * LDB + bcol];
    #pragma unroll
    for (int i = 0; i < 2; ++i) {
      bf16x8 b;
      #pragma unroll
      for (int e = 0; e < 4; ++e) {
        b[e] = (bf16)rb[2 * i][e]; b[4 + e] = (bf16)rb[2 * i + 1][e];
      }
      *(bf16x8*)(lB + i * 8) = b;
    }
  };

  // prologue: stage tile 0
  if constexpr (!A_IS_F32) stageA_async(0, 0);
  stage_load(0);
  stage_store(0);
  if constexpr (!A_IS_F32) wait_asynccnt0();
  __syncthreads();

  const int nK = K / KT;
  for (int kt = 0; kt < nK; ++kt) {
    const int cur = kt & 1;
    const bool more = (kt + 1) < nK;
    if (more) {
      if constexpr (!A_IS_F32) stageA_async((kt + 1) * KT, cur ^ 1);
      stage_load((kt + 1) * KT);
    }
    if (kt + 2 < nK) {
      __builtin_prefetch(gA + (kt + 2) * KT, 0, 3);
      __builtin_prefetch(gB + (kt + 2) * KT, 0, 3);
    }

    #pragma unroll
    for (int ks = 0; ks < 2; ++ks) {
      v16bf aF[2], bF[2];
      #pragma unroll
      for (int mb = 0; mb < 2; ++mb) {
        const bf16* p = &sA[cur][(wM * 32 + mb * 16 + r) * LDA + ks * 32 + koffA];
        aF[mb] = pack16(*(const bf16x8*)p, *(const bf16x8*)(p + 8));
      }
      #pragma unroll
      for (int nb = 0; nb < 2; ++nb) {
        const bf16* p = &sB[cur][(wN * 32 + nb * 16 + r) * LDB + ks * 32 + koffB];
        bF[nb] = pack16(*(const bf16x8*)p, *(const bf16x8*)(p + 8));
      }
      #pragma unroll
      for (int mb = 0; mb < 2; ++mb)
        #pragma unroll
        for (int nb = 0; nb < 2; ++nb)
          acc[mb][nb] = wmma_bf16(aF[mb], bF[nb], acc[mb][nb]);
    }

    if (more) {
      stage_store(cur ^ 1);
      if constexpr (!A_IS_F32) wait_asynccnt0();
    }
    __syncthreads();
  }

  // --- epilogue: bias add + store ---
  #pragma unroll
  for (int mb = 0; mb < 2; ++mb) {
    #pragma unroll
    for (int nb = 0; nb < 2; ++nb) {
      const int col = n0 + wN * 32 + nb * 16 + r;
      const float bv = bias[col];
      #pragma unroll
      for (int rr = 0; rr < 8; ++rr) {
        const int row = m0 + wM * 32 + mb * 16 + rr + hi * 8;
        const float v = acc[mb][nb][rr] + bv;
        if constexpr (sizeof(OT) == 4) C[(size_t)row * N + col] = v;
        else                           C[(size_t)row * N + col] = (bf16)v;
      }
    }
  }
}

// ---------------------------------------------------------------------------
// Flash attention: block = (128-query tile, batch*head). 8 waves, each owns
// 16 query rows. 64-wide key tiles: K tile staged into LDS with async copies
// (shared by all 8 waves), V^T staged transposed, both double-buffered with
// one barrier per tile; softmax row stats in registers with DPP reductions.
// ---------------------------------------------------------------------------
__global__ __launch_bounds__(256) void attn_kernel(
    const bf16* __restrict__ Qp, const bf16* __restrict__ Kp,
    const bf16* __restrict__ Vp, bf16* __restrict__ Ctx)
{
  constexpr int NK   = 64;                  // key-tile width
  constexpr int LDK  = 72;                  // K tile row stride (halves)
  constexpr int LDVT = 72;                  // V^T row stride (halves)
  constexpr int LDP  = 72;                  // P row stride
  __shared__ bf16 sK [2][NK * LDK];         // [64 j][64 d], async-staged
  __shared__ bf16 sVT[2][HEAD_DIM * LDVT];  // [64 d][64 j]
  __shared__ bf16 sP[8 * 16 * LDP];         // per-wave [16 q][64 j]

  const int tid  = threadIdx.x;
  const int lane = tid & 31;
  const int wv   = tid >> 5;
  const int bh   = blockIdx.y;
  const int b    = bh >> 4;
  const int h    = bh & 15;
  const int q0   = blockIdx.x * 128 + wv * 16;

  const int r     = lane & 15;
  const int hi    = lane >> 4;
  const int koffA = hi * 16;
  const int koffB = hi * 16;

  const size_t headBase = ((size_t)b * SEQ) * D_MODEL + (size_t)h * HEAD_DIM;

  // Q fragments (A layout), loaded once: rows q0..q0+15, d = 0..63
  v16bf qF[2];
  {
    const bf16* qp = Qp + headBase + (size_t)(q0 + r) * D_MODEL;
    #pragma unroll
    for (int kk = 0; kk < 2; ++kk) {
      const bf16* p = qp + kk * 32 + koffA;
      qF[kk] = pack16(*(const bf16x8*)p, *(const bf16x8*)(p + 8));
    }
  }

  v8f o[4];
  #pragma unroll
  for (int db = 0; db < 4; ++db)
    o[db] = v8f{0.f, 0.f, 0.f, 0.f, 0.f, 0.f, 0.f, 0.f};
  float mrun[8], lrun[8];
  #pragma unroll
  for (int rr = 0; rr < 8; ++rr) { mrun[rr] = -3.0e38f; lrun[rr] = 0.f; }

  // K staging (async): thread -> (row kj, two 8-half chunks at kc, kc+1)
  const int kj = tid >> 2;
  const int kc = (tid & 3) * 2;
  const bf16* gK = Kp + headBase + (size_t)kj * D_MODEL;
  auto kstage = [&](int j0, int buf) {
    const bf16* g = gK + (size_t)j0 * D_MODEL;
    async_copy_b128(lds_addr(&sK[buf][kj * LDK + kc * 8]),       g + kc * 8);
    async_copy_b128(lds_addr(&sK[buf][kj * LDK + (kc + 1) * 8]), g + (kc + 1) * 8);
  };

  // V^T staging: thread -> (key sj, 16 contiguous d at sd)
  const int sj = tid & 63;
  const int sd = (tid >> 6) * 16;
  const bf16* gV = Vp + headBase + (size_t)sj * D_MODEL + sd;
  auto vload = [&](int j0, bf16x8& a, bf16x8& c) {
    const bf16* g = gV + (size_t)j0 * D_MODEL;
    a = *(const bf16x8*)g;
    c = *(const bf16x8*)(g + 8);
  };
  auto vstore = [&](int buf, bf16x8 a, bf16x8 c) {
    #pragma unroll
    for (int i = 0; i < 8; ++i) sVT[buf][(sd + i) * LDVT + sj] = a[i];
    #pragma unroll
    for (int i = 0; i < 8; ++i) sVT[buf][(sd + 8 + i) * LDVT + sj] = c[i];
  };

  bf16* myP = &sP[wv * 16 * LDP];

  {
    kstage(0, 0);
    bf16x8 va, vc;
    vload(0, va, vc);
    vstore(0, va, vc);
    wait_asynccnt0();
  }
  __syncthreads();

  constexpr int NT = SEQ / NK;              // 32 key tiles
  for (int t = 0; t < NT; ++t) {
    const int cur = t & 1;
    const int j0  = t * NK;
    const bool more = (t + 1) < NT;
    bf16x8 na, nc;
    if (more) {
      kstage(j0 + NK, cur ^ 1);             // async engine runs during compute
      vload(j0 + NK, na, nc);
    }
    if (t + 2 < NT) {
      __builtin_prefetch(gV + (size_t)(j0 + 2 * NK) * D_MODEL, 0, 3);
      __builtin_prefetch(gK + (size_t)(j0 + 2 * NK) * D_MODEL, 0, 3);
    }

    // --- S = Q K^T (K fragments from LDS, shared by all 8 waves) ---
    v8f s[4];
    #pragma unroll
    for (int nb = 0; nb < 4; ++nb)
      s[nb] = v8f{0.f, 0.f, 0.f, 0.f, 0.f, 0.f, 0.f, 0.f};
    #pragma unroll
    for (int nb = 0; nb < 4; ++nb) {
      #pragma unroll
      for (int kk = 0; kk < 2; ++kk) {
        const bf16* kp = &sK[cur][(nb * 16 + r) * LDK + kk * 32 + koffB];
        v16bf kF = pack16(*(const bf16x8*)kp, *(const bf16x8*)(kp + 8));
        s[nb] = wmma_bf16(qF[kk], kF, s[nb]);
      }
    }

    // --- online softmax (per-row stats; DPP16 reductions, no LDS) ---
    float scale[8];
    #pragma unroll
    for (int rr = 0; rr < 8; ++rr) {
      float tm = fmaxf(fmaxf(s[0][rr], s[1][rr]), fmaxf(s[2][rr], s[3][rr]));
      tm = red_max16(tm);
      const float mnew = fmaxf(mrun[rr], tm);
      scale[rr] = __expf(mrun[rr] - mnew);
      mrun[rr] = mnew;
      float ps = 0.f;
      #pragma unroll
      for (int nb = 0; nb < 4; ++nb) {
        const float p = __expf(s[nb][rr] - mnew);
        s[nb][rr] = p;
        ps += p;
      }
      ps = red_add16(ps);
      lrun[rr] = lrun[rr] * scale[rr] + ps;
    }

    // --- rescale O; spill P (C layout -> wave-private LDS) ---
    #pragma unroll
    for (int db = 0; db < 4; ++db)
      #pragma unroll
      for (int rr = 0; rr < 8; ++rr)
        o[db][rr] *= scale[rr];

    #pragma unroll
    for (int nb = 0; nb < 4; ++nb)
      #pragma unroll
      for (int rr = 0; rr < 8; ++rr)
        myP[(rr + hi * 8) * LDP + nb * 16 + r] = (bf16)s[nb][rr];

    // wave-private LDS RAW: order ds stores before ds loads
    asm volatile("s_wait_dscnt 0" ::: "memory");

    // --- O += P @ V ---
    {
      v16bf pF[2];
      #pragma unroll
      for (int kk = 0; kk < 2; ++kk) {
        const bf16* pp = &myP[r * LDP + kk * 32 + koffA];
        pF[kk] = pack16(*(const bf16x8*)pp, *(const bf16x8*)(pp + 8));
      }
      #pragma unroll
      for (int db = 0; db < 4; ++db) {
        #pragma unroll
        for (int kk = 0; kk < 2; ++kk) {
          const bf16* vp = &sVT[cur][(db * 16 + r) * LDVT + kk * 32 + koffB];
          v16bf vF = pack16(*(const bf16x8*)vp, *(const bf16x8*)(vp + 8));
          o[db] = wmma_bf16(pF[kk], vF, o[db]);
        }
      }
    }

    if (more) {
      vstore(cur ^ 1, na, nc);
      wait_asynccnt0();                     // K tile for t+1 landed in LDS
    }
    __syncthreads();
  }

  // --- normalize and store context (bf16) ---
  #pragma unroll
  for (int rr = 0; rr < 8; ++rr) lrun[rr] = 1.0f / lrun[rr];
  #pragma unroll
  for (int db = 0; db < 4; ++db) {
    const int col = h * HEAD_DIM + db * 16 + r;
    #pragma unroll
    for (int rr = 0; rr < 8; ++rr) {
      const int row = q0 + rr + hi * 8;
      Ctx[((size_t)b * SEQ + row) * D_MODEL + col] = (bf16)(o[db][rr] * lrun[rr]);
    }
  }
}

// ---------------------------------------------------------------------------
// Launcher: 3 projection GEMMs -> flash attention -> output GEMM.
// Workspace: Qp|Kp|Vp|Ctx bf16, 4 * 8192*1024*2B = 67 MB, rewritten per call.
// ---------------------------------------------------------------------------
extern "C" void kernel_launch(void* const* d_in, const int* in_sizes, int n_in,
                              void* d_out, int out_size, void* d_ws, size_t ws_size,
                              hipStream_t stream) {
  const float* q  = (const float*)d_in[0];
  const float* k  = (const float*)d_in[1];
  const float* v  = (const float*)d_in[2];
  const float* Wq = (const float*)d_in[3];
  const float* bq = (const float*)d_in[4];
  const float* Wk = (const float*)d_in[5];
  const float* bk = (const float*)d_in[6];
  const float* Wv = (const float*)d_in[7];
  const float* bv = (const float*)d_in[8];
  const float* Wo = (const float*)d_in[9];
  const float* bo = (const float*)d_in[10];
  float* out = (float*)d_out;

  const size_t SZ = (size_t)M_TOKENS * D_MODEL;   // elements
  bf16* Qp  = (bf16*)d_ws;
  bf16* Kp  = Qp + SZ;
  bf16* Vp  = Kp + SZ;
  bf16* Ctx = Vp + SZ;

  dim3 gg(D_MODEL / 64, M_TOKENS / 128);          // (16, 64)
  dim3 blk(256);

  gemm_bias_kernel<float, bf16><<<gg, blk, 0, stream>>>(q, Wq, bq, Qp,
      M_TOKENS, D_MODEL, D_MODEL);
  gemm_bias_kernel<float, bf16><<<gg, blk, 0, stream>>>(k, Wk, bk, Kp,
      M_TOKENS, D_MODEL, D_MODEL);
  gemm_bias_kernel<float, bf16><<<gg, blk, 0, stream>>>(v, Wv, bv, Vp,
      M_TOKENS, D_MODEL, D_MODEL);

  attn_kernel<<<dim3(SEQ / 128, BATCH * N_HEADS), blk, 0, stream>>>(Qp, Kp, Vp, Ctx);

  gemm_bias_kernel<bf16, float><<<gg, blk, 0, stream>>>(Ctx, Wo, bo, out,
      M_TOKENS, D_MODEL, D_MODEL);
}
